// DeformConv2dBlock_35373350650207
// MI455X (gfx1250) — matrix operand order repaired and checked by reading, not compiled, not run on hardware
//
#include <hip/hip_runtime.h>
#include <cstdint>
#include <cstddef>

// ---------------- CDNA5 WMMA types ----------------
typedef __attribute__((ext_vector_type(16))) __bf16 bf16x16;
typedef __attribute__((ext_vector_type(8)))  __bf16 bf16x8;
typedef __attribute__((ext_vector_type(8)))  float  f32x8;

#define BB   4
#define CCH  64
#define HGT  128
#define WID  128
#define OCH  64
#define HWP  (HGT * WID)     // 16384
#define KDIM 576             // C * 9, ordered k2*64 + c
#define WSTR 592             // weight row stride (bf16 elems), 1184B -> 32B aligned rows
#define ASTR 72              // LDS A-panel row stride (bf16 elems), 144B -> 16B aligned

// Compiler scheduling fence (keeps source-order anchoring for memory ops).
#define SCHED_FENCE() asm volatile("" ::: "memory")

__device__ inline f32x8 wmma_bf16(bf16x16 a, bf16x16 b, f32x8 c) {
    // D = A(16x32) * B(32x16) + C, f32 accumulate
    return __builtin_amdgcn_wmma_f32_16x16x32_bf16(false, a, false, b, (short)0, c,
                                                   false, false);
}

// ---------------- Kernel 1: x (NCHW f32) -> xb (B,H,W,C bf16) ----------------
__global__ __launch_bounds__(256) void k_prep_x(const float* __restrict__ x,
                                                __bf16* __restrict__ xb) {
    __shared__ float t[64][33];
    int bid  = blockIdx.x;
    int b    = bid >> 9;               // HW/32 = 512 tiles per batch
    int pix0 = (bid & 511) << 5;       // 32 pixels per tile
    int tid  = threadIdx.x;
    const float* xbase = x + (size_t)b * CCH * HWP;
    #pragma unroll
    for (int i = 0; i < 8; ++i) {
        int l = tid + (i << 8);
        int c = l >> 5, p = l & 31;
        t[c][p] = xbase[(size_t)c * HWP + pix0 + p];   // coalesced per channel row
    }
    __syncthreads();
    __bf16* obase = xb + ((size_t)b * HWP + pix0) * 64;
    #pragma unroll
    for (int i = 0; i < 8; ++i) {
        int l = tid + (i << 8);
        int p = l >> 6, c = l & 63;
        obase[p * 64 + c] = (__bf16)t[c][p];           // contiguous 64-ch rows
    }
}

// ---------------- Kernel 2: repack weights to bf16, zero BN accumulators ------
__global__ __launch_bounds__(256) void k_prep_w(const float* __restrict__ weight,
                                                const float* __restrict__ offset_w,
                                                __bf16* __restrict__ wbf,
                                                __bf16* __restrict__ owbf,
                                                float* __restrict__ bnsum,
                                                float* __restrict__ bnsq) {
    int tid = threadIdx.x;
    for (int i = tid; i < 32 * WSTR; i += 256) owbf[i] = (__bf16)0.f;
    for (int i = tid; i < 64 * WSTR; i += 256) wbf[i]  = (__bf16)0.f;
    __syncthreads();
    // reference flattens weight[o][c][k2] (c outer); we reorder K = k2*64 + c
    for (int i = tid; i < OCH * KDIM; i += 256) {
        int o = i / KDIM, r = i - o * KDIM;
        int c = r / 9,    k2 = r - c * 9;
        wbf[o * WSTR + k2 * 64 + c] = (__bf16)weight[i];
    }
    for (int i = tid; i < 27 * KDIM; i += 256) {
        int n = i / KDIM, r = i - n * KDIM;
        int c = r / 9,    k2 = r - c * 9;
        owbf[n * WSTR + k2 * 64 + c] = (__bf16)offset_w[i];
    }
    if (tid < 64) { bnsum[tid] = 0.f; bnsq[tid] = 0.f; }
}

// ---------------- Kernel 3: offset/mask conv via WMMA implicit GEMM ----------
// M-tile = 64 pixels (one row segment), N = 32 (27 real), K = 576 (9 taps x 64c)
// A-panel staged with async global->LDS copies (ASYNCcnt), double-buffered.
__global__ __launch_bounds__(256) void k_offset_conv(const __bf16* __restrict__ xb,
                                                     const __bf16* __restrict__ owbf,
                                                     const float* __restrict__ offset_b,
                                                     float* __restrict__ om) {
    __shared__ __bf16 ldsA[2][64 * ASTR];
    int tid  = threadIdx.x;
    int wv   = tid >> 5, lane = tid & 31;
    int mblk = wv & 3,   nblk = wv >> 2;          // 4 m-blocks x 2 n-blocks
    int bid  = blockIdx.x;
    int b    = bid >> 8;
    int pix0 = (bid & 255) << 6;                   // 64 consecutive pixels (same h)
    int h    = pix0 >> 7, w0 = pix0 & 127;
    const __bf16* xbb = xb + (size_t)b * HWP * 64;
    int pl = tid >> 2, cpart = (tid & 3) << 4;     // 4 threads per pixel, 16 ch each
    int mrow = mblk * 16 + (lane & 15);
    int kh8  = (lane >> 4) << 3;                   // unused; A reads use kh16 (0/16)
    int nrow = nblk * 16 + (lane & 15);
    int kh16 = (lane >> 4) << 4;                   // B lane k-offset (0/16)
    f32x8 acc = {0.f, 0.f, 0.f, 0.f, 0.f, 0.f, 0.f, 0.f};

    auto issue_tap = [&](int k2, int buf) {
        int ki = k2 / 3, kj = k2 - ki * 3;
        int y = h + ki - 1;
        int xcol = w0 + pl + kj - 1;
        __bf16* dst = &ldsA[buf][pl * ASTR + cpart];
        if (y >= 0 && y < HGT && xcol >= 0 && xcol < WID) {
            // pure copy: async global -> LDS (16B x2 per thread), no VGPR round-trip
            const __bf16* g = xbb + (((size_t)y << 7) + xcol) * 64 + cpart;
            uint32_t ldso = (uint32_t)(uintptr_t)dst;     // addr[31:0] == LDS offset
            asm volatile("global_load_async_to_lds_b128 %0, %1, off"
                         :: "v"(ldso), "v"(g) : "memory");
            asm volatile("global_load_async_to_lds_b128 %0, %1, off"
                         :: "v"(ldso + 16u), "v"(g + 8) : "memory");
        } else {
            bf16x8 z;
            #pragma unroll
            for (int j = 0; j < 8; ++j) z[j] = (__bf16)0.f;
            *(bf16x8*)dst       = z;
            *(bf16x8*)(dst + 8) = z;
        }
    };

    issue_tap(0, 0);
    for (int k2 = 0; k2 < 9; ++k2) {
        int buf = k2 & 1;
        asm volatile("s_wait_asynccnt 0x0" ::: "memory");  // our async copies done
        __syncthreads();                                    // panel visible WGP-wide
        SCHED_FENCE();
        if (k2 < 8) issue_tap(k2 + 1, buf ^ 1);             // overlap next tap
        #pragma unroll
        for (int step = 0; step < 2; ++step) {
            int ko = step << 5;
            bf16x8 alo = *(const bf16x8*)&ldsA[buf][mrow * ASTR + ko + kh16];
            bf16x8 ahi = *(const bf16x8*)&ldsA[buf][mrow * ASTR + ko + kh16 + 8];
            bf16x16 afrag;
            #pragma unroll
            for (int j = 0; j < 8; ++j) { afrag[j] = alo[j]; afrag[j + 8] = ahi[j]; }
            bf16x16 bfrag = *(const bf16x16*)(owbf + nrow * WSTR + k2 * 64 + ko + kh16);
            acc = wmma_bf16(afrag, bfrag, acc);
        }
    }
    if (nrow < 27) {
        float ob = offset_b[nrow];
        #pragma unroll
        for (int r = 0; r < 8; ++r) {
            int m   = r + ((lane >> 4) << 3);
            int pix = pix0 + mblk * 16 + m;
            om[((size_t)b * HWP + pix) * 32 + nrow] = acc[r] + ob;
        }
    }
}

// ---------------- Kernel 4: sampling metadata --------------------------------
__global__ __launch_bounds__(256) void k_meta(const float* __restrict__ om,
                                              uint32_t* __restrict__ midx,
                                              float4* __restrict__ mw) {
    int total = BB * 9 * HWP;
    for (int s = blockIdx.x * 256 + threadIdx.x; s < total; s += gridDim.x * 256) {
        int pix = s & (HWP - 1);
        int t   = s >> 14;
        int k2  = t % 9;
        int b   = t / 9;
        const float* omp = om + ((size_t)b * HWP + pix) * 32;
        float dy = omp[k2], dx = omp[9 + k2];
        float mval = 1.f / (1.f + __expf(-omp[18 + k2]));
        int hh = pix >> 7, ww = pix & 127;
        int ki = k2 / 3, kj = k2 - ki * 3;
        float ys = (float)(hh - 1 + ki) + dy;
        float xs = (float)(ww - 1 + kj) + dx;
        float y0f = floorf(ys), x0f = floorf(xs);
        float wy1 = ys - y0f, wx1 = xs - x0f;
        float wy0 = 1.f - wy1, wx0 = 1.f - wx1;
        int y0 = (int)y0f, x0 = (int)x0f;
        int y1 = y0 + 1,   x1 = x0 + 1;
        float vy0 = (y0 >= 0 && y0 < HGT) ? 1.f : 0.f;
        float vy1 = (y1 >= 0 && y1 < HGT) ? 1.f : 0.f;
        float vx0 = (x0 >= 0 && x0 < WID) ? 1.f : 0.f;
        float vx1 = (x1 >= 0 && x1 < WID) ? 1.f : 0.f;
        int yA = min(max(y0, 0), HGT - 1), yB = min(max(y1, 0), HGT - 1);
        int xA = min(max(x0, 0), WID - 1), xB = min(max(x1, 0), WID - 1);
        float4 wgt;
        wgt.x = wy0 * wx0 * mval * vy0 * vx0;
        wgt.y = wy0 * wx1 * mval * vy0 * vx1;
        wgt.z = wy1 * wx0 * mval * vy1 * vx0;
        wgt.w = wy1 * wx1 * mval * vy1 * vx1;
        midx[s] = (uint32_t)yA | ((uint32_t)xA << 8) | ((uint32_t)yB << 16) |
                  ((uint32_t)xB << 24);
        mw[s] = wgt;
    }
}

// ---------------- Kernel 5: deformable main conv via WMMA --------------------
// M-tile 64 x N 64, K = 576; A-panel built per-tap by fused bilinear gather,
// double-buffered so tap k+1's gather overlaps tap k's WMMAs.
__global__ __launch_bounds__(256) void k_main_conv(const __bf16* __restrict__ xb,
                                                   const __bf16* __restrict__ wbf,
                                                   const uint32_t* __restrict__ midx,
                                                   const float4* __restrict__ mw,
                                                   const float* __restrict__ bias,
                                                   float* __restrict__ convout,
                                                   float* __restrict__ bnsum,
                                                   float* __restrict__ bnsq) {
    __shared__ __bf16 ldsA[2][64 * ASTR];
    __shared__ float  ldsSum[64];
    __shared__ float  ldsSq[64];
    int tid = threadIdx.x;
    if (tid < 64) { ldsSum[tid] = 0.f; ldsSq[tid] = 0.f; }
    int wv    = tid >> 5, lane = tid & 31;
    int mblk  = wv & 3;
    int nblk0 = (wv >> 2) << 1;                    // each wave owns 2 n-blocks
    int bid   = blockIdx.x;
    int b     = bid >> 8;
    int pix0  = (bid & 255) << 6;
    const __bf16* xbb = xb + (size_t)b * HWP * 64;
    int pl = tid >> 2, cpart = (tid & 3) << 4;
    int mrow = mblk * 16 + (lane & 15);
    int kh8  = (lane >> 4) << 3;
    int nlo  = lane & 15;
    int kh16 = (lane >> 4) << 4;
    f32x8 acc0 = {0.f, 0.f, 0.f, 0.f, 0.f, 0.f, 0.f, 0.f};
    f32x8 acc1 = {0.f, 0.f, 0.f, 0.f, 0.f, 0.f, 0.f, 0.f};

    auto gather_tap = [&](int k2, int buf) {
        int s = (b * 9 + k2) * HWP + pix0 + pl;
        uint32_t mm = midx[s];
        float4 wg = mw[s];
        int yA = mm & 255, xA = (mm >> 8) & 255;
        int yB = (mm >> 16) & 255, xB = (mm >> 24) & 255;
        bf16x16 a00 = *(const bf16x16*)(xbb + (((size_t)yA << 7) + xA) * 64 + cpart);
        bf16x16 a01 = *(const bf16x16*)(xbb + (((size_t)yA << 7) + xB) * 64 + cpart);
        bf16x16 a10 = *(const bf16x16*)(xbb + (((size_t)yB << 7) + xA) * 64 + cpart);
        bf16x16 a11 = *(const bf16x16*)(xbb + (((size_t)yB << 7) + xB) * 64 + cpart);
        bf16x8 rlo, rhi;
        #pragma unroll
        for (int j = 0; j < 8; ++j) {
            rlo[j] = (__bf16)(wg.x * (float)a00[j]     + wg.y * (float)a01[j] +
                              wg.z * (float)a10[j]     + wg.w * (float)a11[j]);
            rhi[j] = (__bf16)(wg.x * (float)a00[j + 8] + wg.y * (float)a01[j + 8] +
                              wg.z * (float)a10[j + 8] + wg.w * (float)a11[j + 8]);
        }
        *(bf16x8*)&ldsA[buf][pl * ASTR + cpart]     = rlo;
        *(bf16x8*)&ldsA[buf][pl * ASTR + cpart + 8] = rhi;
    };

    gather_tap(0, 0);
    for (int k2 = 0; k2 < 9; ++k2) {
        int buf = k2 & 1;
        __syncthreads();                         // buf's stores visible WGP-wide
        SCHED_FENCE();
        if (k2 < 8) gather_tap(k2 + 1, buf ^ 1); // overlap next tap's gather
        #pragma unroll
        for (int step = 0; step < 2; ++step) {
            int ko = step << 5;
            bf16x8 alo = *(const bf16x8*)&ldsA[buf][mrow * ASTR + ko + kh16];
            bf16x8 ahi = *(const bf16x8*)&ldsA[buf][mrow * ASTR + ko + kh16 + 8];
            bf16x16 afrag;
            #pragma unroll
            for (int j = 0; j < 8; ++j) { afrag[j] = alo[j]; afrag[j + 8] = ahi[j]; }
            int kg = k2 * 64 + ko + kh16;
            bf16x16 b0 = *(const bf16x16*)(wbf + (nblk0 * 16 + nlo) * WSTR + kg);
            bf16x16 b1 = *(const bf16x16*)(wbf + ((nblk0 + 1) * 16 + nlo) * WSTR + kg);
            acc0 = wmma_bf16(afrag, b0, acc0);
            acc1 = wmma_bf16(afrag, b1, acc1);
        }
        if (k2 < 8)  // warm L0/L2 for next tap's weight slab (global_prefetch_b8)
            __builtin_prefetch(wbf + (nblk0 * 16 + nlo) * WSTR + (k2 + 1) * 64, 0, 2);
    }
    __syncthreads();
    SCHED_FENCE();
    // epilogue: bias, store channel-last, BN partial sums (LDS then global atomics)
    int ch0 = nblk0 * 16 + nlo, ch1 = ch0 + 16;
    float bv0 = bias[ch0], bv1 = bias[ch1];
    float s0 = 0.f, q0 = 0.f, s1 = 0.f, q1 = 0.f;
    #pragma unroll
    for (int r = 0; r < 8; ++r) {
        int m = r + ((lane >> 4) << 3);
        size_t row = ((size_t)b * HWP + pix0 + mblk * 16 + m) * 64;
        float v0 = acc0[r] + bv0;
        float v1 = acc1[r] + bv1;
        convout[row + ch0] = v0;
        convout[row + ch1] = v1;
        s0 += v0; q0 += v0 * v0;
        s1 += v1; q1 += v1 * v1;
    }
    atomicAdd(&ldsSum[ch0], s0); atomicAdd(&ldsSq[ch0], q0);
    atomicAdd(&ldsSum[ch1], s1); atomicAdd(&ldsSq[ch1], q1);
    __syncthreads();
    SCHED_FENCE();
    if (tid < 64) {
        atomicAdd(&bnsum[tid], ldsSum[tid]);
        atomicAdd(&bnsq[tid],  ldsSq[tid]);
    }
}

// ---------------- Kernel 6: BN finalize --------------------------------------
__global__ void k_bnfin(const float* __restrict__ bnsum, const float* __restrict__ bnsq,
                        const float* __restrict__ gamma, const float* __restrict__ beta,
                        float* __restrict__ bnparam) {
    int c = threadIdx.x;
    if (c < 64) {
        float n    = (float)(BB * HWP);
        float mean = bnsum[c] / n;
        float var  = bnsq[c] / n - mean * mean;
        float sc   = gamma[c] * rsqrtf(var + 1e-5f);
        bnparam[c]      = sc;
        bnparam[64 + c] = beta[c] - mean * sc;
    }
}

// ---------------- Kernel 7: BN apply + ReLU, transpose back to NCHW ----------
__global__ __launch_bounds__(256) void k_apply(const float* __restrict__ convout,
                                               const float* __restrict__ bnparam,
                                               float* __restrict__ out) {
    __shared__ float t[64 * 65];
    int bid  = blockIdx.x;
    int b    = bid >> 8;
    int pix0 = (bid & 255) << 6;
    int tid  = threadIdx.x;
    const float* src = convout + ((size_t)b * HWP + pix0) * 64;
    #pragma unroll
    for (int i = 0; i < 16; ++i) {
        int l = tid + (i << 8);
        int p = l >> 6, ch = l & 63;
        t[ch * 65 + p] = src[l];               // coalesced read, transposed LDS store
    }
    __syncthreads();
    float* ob = out + (size_t)b * 64 * HWP + pix0;
    #pragma unroll
    for (int i = 0; i < 16; ++i) {
        int l = tid + (i << 8);
        int ch = l >> 6, p = l & 63;
        float v = t[ch * 65 + p] * bnparam[ch] + bnparam[64 + ch];
        ob[(size_t)ch * HWP + p] = fmaxf(v, 0.f);   // coalesced NCHW write
    }
}

// ---------------- host-side launcher -----------------------------------------
extern "C" void kernel_launch(void* const* d_in, const int* in_sizes, int n_in,
                              void* d_out, int out_size, void* d_ws, size_t ws_size,
                              hipStream_t stream) {
    (void)in_sizes; (void)n_in; (void)out_size; (void)ws_size;
    const float* x        = (const float*)d_in[0];
    const float* offset_w = (const float*)d_in[1];
    const float* offset_b = (const float*)d_in[2];
    const float* weight   = (const float*)d_in[3];
    const float* bias     = (const float*)d_in[4];
    const float* gamma    = (const float*)d_in[5];
    const float* beta     = (const float*)d_in[6];

    char* ws = (char*)d_ws;
    size_t off = 0;
    auto carve = [&](size_t bytes) -> void* {
        void* p = ws + off;
        off = (off + bytes + 255) & ~(size_t)255;
        return p;
    };
    __bf16*   xb      = (__bf16*)  carve((size_t)BB * HWP * 64 * 2);       // 8 MB
    __bf16*   wbf     = (__bf16*)  carve((size_t)64 * WSTR * 2);
    __bf16*   owbf    = (__bf16*)  carve((size_t)32 * WSTR * 2);
    float*    om      = (float*)   carve((size_t)BB * HWP * 32 * 4);       // 8 MB
    uint32_t* midx    = (uint32_t*)carve((size_t)BB * 9 * HWP * 4);        // 2.25 MB
    float4*   mw      = (float4*)  carve((size_t)BB * 9 * HWP * 16);       // 9 MB
    float*    convout = (float*)   carve((size_t)BB * HWP * 64 * 4);       // 16 MB
    float*    bnsum   = (float*)   carve(64 * 4);
    float*    bnsq    = (float*)   carve(64 * 4);
    float*    bnparam = (float*)   carve(128 * 4);
    float*    outp    = (float*)d_out;

    k_prep_x     <<<BB * (HWP / 32), 256, 0, stream>>>(x, xb);
    k_prep_w     <<<1, 256, 0, stream>>>(weight, offset_w, wbf, owbf, bnsum, bnsq);
    k_offset_conv<<<BB * (HWP / 64), 256, 0, stream>>>(xb, owbf, offset_b, om);
    k_meta       <<<(BB * 9 * HWP + 255) / 256, 256, 0, stream>>>(om, midx, mw);
    k_main_conv  <<<BB * (HWP / 64), 256, 0, stream>>>(xb, wbf, midx, mw, bias,
                                                       convout, bnsum, bnsq);
    k_bnfin      <<<1, 64, 0, stream>>>(bnsum, bnsq, gamma, beta, bnparam);
    k_apply      <<<BB * (HWP / 64), 256, 0, stream>>>(convout, bnparam, outp);
}
